// FullRecContract_10101763080618
// MI455X (gfx1250) — matrix operand order repaired and checked by reading, not compiled, not run on hardware
//
#include <hip/hip_runtime.h>

#define RF 500000
#define F1 256
#define NF 128
#define RA 50000
#define MTILE 80           // rows per block (500000 = 80 * 6250, no tail)
#define NSUB (MTILE / 16)  // 5 M-subtiles per wave

typedef __bf16 bf16;
typedef __attribute__((ext_vector_type(16))) __bf16 v16bf;
typedef __attribute__((ext_vector_type(8)))  __bf16 v8bf;
typedef __attribute__((ext_vector_type(8)))  float  v8f;

// ---------- helpers ----------
static __device__ inline unsigned short f2bf_bits(float f) {
  unsigned u = __float_as_uint(f);
  u += 0x7fffu + ((u >> 16) & 1u);          // round-to-nearest-even
  return (unsigned short)(u >> 16);
}
static __device__ inline float bf_bits2f(unsigned short h) {
  return __uint_as_float(((unsigned)h) << 16);
}
static __device__ inline bf16 bits2bf(unsigned short h) {
  return __builtin_bit_cast(bf16, h);
}
// order-preserving float<->uint key for atomicMax over signed floats
static __device__ inline unsigned floatKey(float x) {
  unsigned u = __float_as_uint(x);
  return (u & 0x80000000u) ? ~u : (u | 0x80000000u);
}
static __device__ inline float keyToFloat(unsigned k) {
  return (k & 0x80000000u) ? __uint_as_float(k & 0x7fffffffu)
                           : __uint_as_float(~k);
}
// unpack 8 packed-bf16-pair dwords into a v16bf B fragment
static __device__ inline v16bf unpackB(uint4 a, uint4 b) {
  unsigned d[8] = {a.x, a.y, a.z, a.w, b.x, b.y, b.z, b.w};
  v16bf o;
#pragma unroll
  for (int i = 0; i < 8; ++i) {
    o[2 * i]     = bits2bf((unsigned short)(d[i] & 0xffffu));
    o[2 * i + 1] = bits2bf((unsigned short)(d[i] >> 16));
  }
  return o;
}

// ---------- kernel 1: zero output / segment sums / global-max key ----------
__global__ void frc_zero(float* __restrict__ out, float* __restrict__ segsum,
                         unsigned* __restrict__ gkey) {
  int i = blockIdx.x * 256 + threadIdx.x;
  if (i < RA * NF) out[i] = 0.f;
  if (i < RA) segsum[i] = 0.f;
  if (i == 0) *gkey = 0u;                   // smallest possible key
}

// ---------- kernel 2: pack W_feat into WMMA-B fragment layout (bf16 hi/lo) --
// Per-lane-contiguous layout: Wf[((kb*2 + khalf)*NF + n)*8 + v] holds the
// packed bf16 pair for K = kb*32 + khalf*16 + 2v (and +1), column n. A lane's
// whole fragment is 8 consecutive dwords -> two global_load_b128.
__global__ void frc_wconv(const float* __restrict__ W,
                          unsigned* __restrict__ WfHi,
                          unsigned* __restrict__ WfLo) {
  int i = blockIdx.x * 256 + threadIdx.x;
  if (i >= (F1 / 2) * NF) return;           // 16384 packed entries
  int v = i & 7;
  int n = (i >> 3) & (NF - 1);
  int kbh = i >> 10;                        // 0..15 = kb*2 + khalf
  int kb = kbh >> 1, khalf = kbh & 1;
  int k0 = kb * 32 + khalf * 16 + 2 * v;
  float w0 = W[(size_t)k0 * NF + n];
  float w1 = W[(size_t)(k0 + 1) * NF + n];
  unsigned short h0 = f2bf_bits(w0), h1 = f2bf_bits(w1);
  unsigned short l0 = f2bf_bits(w0 - bf_bits2f(h0));
  unsigned short l1 = f2bf_bits(w1 - bf_bits2f(h1));
  WfHi[i] = (unsigned)h0 | ((unsigned)h1 << 16);
  WfLo[i] = (unsigned)l0 | ((unsigned)l1 << 16);
}

// ---------- kernel 3: attention logits + global max (one row per wave) -----
__global__ __launch_bounds__(256) void frc_logits(
    const float* __restrict__ x, const float* __restrict__ W_atn,
    const float* __restrict__ b_atn, float* __restrict__ L,
    unsigned* __restrict__ gkey) {
  const int wave = threadIdx.x >> 5;
  const int lane = threadIdx.x & 31;
  const int r = blockIdx.x * 8 + wave;
  const float* xr = x + (size_t)r * F1;
  float s = 0.f;
#pragma unroll
  for (int j = 0; j < F1 / 32; ++j) {
    float v = xr[lane + j * 32];
    v = v > 0.f ? v : 0.01f * v;            // leaky_relu
    s += v * W_atn[lane + j * 32];
  }
#pragma unroll
  for (int m = 1; m <= 16; m <<= 1) s += __shfl_xor(s, m, 32);
  s += b_atn[0];
  __shared__ float red[8];
  if (lane == 0) { L[r] = s; red[wave] = s; }
  __syncthreads();
  if (threadIdx.x == 0) {
    float mx = red[0];
#pragma unroll
    for (int i = 1; i < 8; ++i) mx = fmaxf(mx, red[i]);
    atomicMax(gkey, floatKey(mx));
  }
}

// ---------- kernel 4: exp(L - max) + segment denominators ----------
__global__ void frc_expseg(const float* __restrict__ L,
                           const unsigned* __restrict__ gkey,
                           const int* __restrict__ seg,
                           float* __restrict__ E, float* __restrict__ segsum) {
  int r = blockIdx.x * 256 + threadIdx.x;
  if (r >= RF) return;
  float gm = keyToFloat(*gkey);
  float e = expf(L[r] - gm);
  E[r] = e;
  atomicAdd(&segsum[seg[r]], e);
}

// ---------- kernel 5: WMMA GEMM (bf16x3) + bias + LayerNorm + pooled scatter
// Block: 256 threads = 8 waves; tile = 80 rows x 128 cols. Each wave owns one
// 16-col N-tile across 5 M-subtiles (5 accumulators). Per K-block: one B
// hi/lo fragment (4x global_load_b128) feeds 15 WMMAs.
__global__ __launch_bounds__(256) void frc_gemm_ln_pool(
    const float* __restrict__ x, const unsigned* __restrict__ WfHi,
    const unsigned* __restrict__ WfLo, const float* __restrict__ b_feat,
    const float* __restrict__ gamma, const float* __restrict__ beta,
    const float* __restrict__ E, const float* __restrict__ segsum,
    const int* __restrict__ seg, float* __restrict__ out) {
  __shared__ unsigned short sAhi[MTILE * F1];  // 40 KB bf16 hi tile
  __shared__ unsigned short sAlo[MTILE * F1];  // 40 KB bf16 lo tile
  __shared__ float sFeat[MTILE * NF];          // 40 KB f32 feat tile

  const int tid = threadIdx.x;
  const int row0 = blockIdx.x * MTILE;

  // Stage 80x256 activations: f32 load -> leaky_relu -> bf16 hi/lo in LDS.
  const float4* xr = (const float4*)x + (size_t)row0 * (F1 / 4);
#pragma unroll
  for (int it = 0; it < (MTILE * F1 / 4) / 256; ++it) {  // 20 float4/thread
    int i = tid + it * 256;
    float4 v = xr[i];
    float av[4] = {v.x, v.y, v.z, v.w};
#pragma unroll
    for (int j = 0; j < 4; ++j) {
      float a = av[j] > 0.f ? av[j] : 0.01f * av[j];
      unsigned short hi = f2bf_bits(a);
      sAhi[i * 4 + j] = hi;
      sAlo[i * 4 + j] = f2bf_bits(a - bf_bits2f(hi));
    }
  }
  __syncthreads();

  const int wave = tid >> 5;
  const int lane = tid & 31;
  const int m16 = lane & 15;
  const int khalf = lane >> 4;
  const int ncol = wave * 16 + m16;         // B/N column this lane supplies

  v8f acc[NSUB];
#pragma unroll
  for (int t = 0; t < NSUB; ++t) acc[t] = (v8f){};

#pragma unroll
  for (int kb = 0; kb < F1 / 32; ++kb) {
    // B fragments: 8 contiguous dwords per lane -> 2x b128 each for hi/lo.
    const unsigned* bh = WfHi + ((size_t)(kb * 2 + khalf) * NF + ncol) * 8;
    const unsigned* bl = WfLo + ((size_t)(kb * 2 + khalf) * NF + ncol) * 8;
    const v16bf bHi = unpackB(*(const uint4*)bh, *(const uint4*)(bh + 4));
    const v16bf bLo = unpackB(*(const uint4*)bl, *(const uint4*)(bl + 4));

    const int k0 = kb * 32 + khalf * 16;    // A-frag K chunks: k0, k0+8
#pragma unroll
    for (int t = 0; t < NSUB; ++t) {
      const int mrow = t * 16 + m16;
      v8bf ah0 = *(const v8bf*)(const void*)&sAhi[mrow * F1 + k0];
      v8bf ah1 = *(const v8bf*)(const void*)&sAhi[mrow * F1 + k0 + 8];
      v8bf al0 = *(const v8bf*)(const void*)&sAlo[mrow * F1 + k0];
      v8bf al1 = *(const v8bf*)(const void*)&sAlo[mrow * F1 + k0 + 8];
      v16bf aHi, aLo;
#pragma unroll
      for (int i = 0; i < 8; ++i) {
        aHi[i] = ah0[i]; aHi[8 + i] = ah1[i];
        aLo[i] = al0[i]; aLo[8 + i] = al1[i];
      }
      acc[t] = __builtin_amdgcn_wmma_f32_16x16x32_bf16(
          false, aHi, false, bHi, (short)0, acc[t], false, false);
      acc[t] = __builtin_amdgcn_wmma_f32_16x16x32_bf16(
          false, aLo, false, bHi, (short)0, acc[t], false, false);
      acc[t] = __builtin_amdgcn_wmma_f32_16x16x32_bf16(
          false, aHi, false, bLo, (short)0, acc[t], false, false);
    }
  }

  // C/D layout: lanes 0-15 -> M=j, lanes 16-31 -> M=j+8; N = lane&15.
  {
    const int cn = wave * 16 + (lane & 15);
    const int mb = (lane < 16) ? 0 : 8;
    const float bias = b_feat[cn];
#pragma unroll
    for (int t = 0; t < NSUB; ++t)
#pragma unroll
      for (int j = 0; j < 8; ++j)
        sFeat[(t * 16 + mb + j) * NF + cn] = acc[t][j] + bias;
  }
  __syncthreads();

  // LayerNorm per row (128 cols) + softmax-weighted scatter-add.
  const int sub = tid & 15;                 // 16 threads per row, 8 cols each
#pragma unroll
  for (int it = 0; it < NSUB; ++it) {
    const int row = it * 16 + (tid >> 4);
    const float* rp = &sFeat[row * NF + sub * 8];
    float s = 0.f;
#pragma unroll
    for (int j = 0; j < 8; ++j) s += rp[j];
#pragma unroll
    for (int m = 1; m <= 8; m <<= 1) s += __shfl_xor(s, m, 32);
    const float mean = s * (1.0f / NF);
    float vs = 0.f;
#pragma unroll
    for (int j = 0; j < 8; ++j) { float d = rp[j] - mean; vs += d * d; }
#pragma unroll
    for (int m = 1; m <= 8; m <<= 1) vs += __shfl_xor(vs, m, 32);
    const float rstd = rsqrtf(vs * (1.0f / NF) + 1e-5f);

    const int r = row0 + row;
    const int sidx = seg[r];
    const float w = E[r] / segsum[sidx];
    float* op = out + (size_t)sidx * NF + sub * 8;
#pragma unroll
    for (int j = 0; j < 8; ++j) {
      const int c = sub * 8 + j;
      const float y = (rp[j] - mean) * rstd * gamma[c] + beta[c];
      atomicAdd(&op[j], y * w);
    }
  }
}

// ---------- launcher ----------
extern "C" void kernel_launch(void* const* d_in, const int* in_sizes, int n_in,
                              void* d_out, int out_size, void* d_ws,
                              size_t ws_size, hipStream_t stream) {
  (void)in_sizes; (void)n_in; (void)out_size; (void)ws_size;
  const float* x      = (const float*)d_in[0];
  const float* W_feat = (const float*)d_in[1];
  const float* b_feat = (const float*)d_in[2];
  const float* gamma  = (const float*)d_in[3];
  const float* beta   = (const float*)d_in[4];
  const float* W_atn  = (const float*)d_in[5];
  const float* b_atn  = (const float*)d_in[6];
  const int*   seg    = (const int*)d_in[7];
  float* out = (float*)d_out;

  // workspace layout (floats): L[RF] | E[RF] | segsum[RA] | gkey | WfHi | WfLo
  // total ~4.33 MB
  float* ws = (float*)d_ws;
  float* L = ws;
  float* E = ws + RF;
  float* segsum = ws + 2 * RF;
  unsigned* gkey = (unsigned*)(ws + 2 * RF + RA);
  unsigned* WfHi = gkey + 1;
  unsigned* WfLo = WfHi + (F1 / 2) * NF;

  frc_zero<<<(RA * NF + 255) / 256, 256, 0, stream>>>(out, segsum, gkey);
  frc_wconv<<<((F1 / 2) * NF + 255) / 256, 256, 0, stream>>>(W_feat, WfHi, WfLo);
  frc_logits<<<RF / 8, 256, 0, stream>>>(x, W_atn, b_atn, L, gkey);
  frc_expseg<<<(RF + 255) / 256, 256, 0, stream>>>(L, gkey, seg, E, segsum);
  frc_gemm_ln_pool<<<RF / MTILE, 256, 0, stream>>>(x, WfHi, WfLo, b_feat,
                                                   gamma, beta, E, segsum,
                                                   seg, out);
}
